// SparseSelfAttention_49976239456740
// MI455X (gfx1250) — matrix operand order repaired and checked by reading, not compiled, not run on hardware
//
#include <hip/hip_runtime.h>
#include <hip/hip_bf16.h>
#include <math.h>

#define DIM 512
#define NHEAD 8
#define DK 64

typedef __attribute__((ext_vector_type(16))) __bf16 v16bf;
typedef __attribute__((ext_vector_type(8)))  __bf16 v8bf;
typedef __attribute__((ext_vector_type(8)))  float  v8f;

__device__ __forceinline__ __bf16 f2bf(float f) {
  union { float f; unsigned u; } in; in.f = f;
  unsigned r = in.u + 0x7FFFu + ((in.u >> 16) & 1u);   // round-to-nearest-even
  unsigned short s = (unsigned short)(r >> 16);
  union { unsigned short s; __bf16 b; } out; out.s = s;
  return out.b;
}

#define CAT16(lo, hi) __builtin_shufflevector(lo, hi, 0,1,2,3,4,5,6,7,8,9,10,11,12,13,14,15)

// ---------------- conversion / prep ----------------

__global__ void cvt_f32_bf16(const float* __restrict__ src, __bf16* __restrict__ dst, int n) {
  int i = blockIdx.x * blockDim.x + threadIdx.x;
  if (i < n) dst[i] = f2bf(src[i]);
}

// WT: 2048 x 512 bf16, row r = output-column r of [Wq|Wk|Wv|Wo], contiguous in k.
// biasAll: 2048 floats = [bq|bk|bv|bo].
__global__ void prep_weights(const float* __restrict__ Wq, const float* __restrict__ Wk,
                             const float* __restrict__ Wv, const float* __restrict__ Wo,
                             const float* __restrict__ bq, const float* __restrict__ bk,
                             const float* __restrict__ bv, const float* __restrict__ bo,
                             __bf16* __restrict__ WT, float* __restrict__ biasAll) {
  int idx = blockIdx.x * blockDim.x + threadIdx.x;
  if (idx >= 2048 * DIM) return;
  int r = idx / DIM;   // fused output column
  int k = idx % DIM;
  const float* W; const float* b; int c;
  if (r < 512)       { W = Wq; b = bq; c = r; }
  else if (r < 1024) { W = Wk; b = bk; c = r - 512; }
  else if (r < 1536) { W = Wv; b = bv; c = r - 1024; }
  else               { W = Wo; b = bo; c = r - 1536; }
  WT[(size_t)r * DIM + k] = f2bf(W[(size_t)k * DIM + c]);
  if (k == 0) biasAll[r] = b[c];
}

// ---------------- bf16 WMMA GEMM (K = DIM = 512 baked in) ----------------
// C[M x Ncols] (fp32, ldc) = A[M x 512](bf16, row-major) * BT^T + bias
// BT is [Ncols x 512] bf16: row n holds column n of B, contiguous in k.
// Block = 128 threads = 4 waves; wave tile = 32(M) x 64(N) -> 8 accumulators;
// block tile 128x64. k-loop fully unrolled: scheduler hoists loads across steps.
__global__ __launch_bounds__(128)
void gemm_bf16_wmma(const __bf16* __restrict__ A,
                    const __bf16* __restrict__ BT,
                    const float* __restrict__ bias,
                    float* __restrict__ C, int ldc,
                    int M) {
  const int lane = threadIdx.x & 31;
  const int wave = threadIdx.x >> 5;
  const int mr = blockIdx.x * 128 + wave * 32;   // this wave: rows mr..mr+31
  const int nc = blockIdx.y * 64;

  // A layout: lane = M (0..15); lanes 0-15 hold K 0..15, 16-31 hold K 16..31;
  // vector elems 0..15 -> K = khA + i.
  const int mrow = lane & 15;
  const int khA  = (lane >> 4) * 16;
  // B layout (same k order as A): lane = N (0..15); lanes 0-15 hold K 0..15, 16-31 hold K 16..31.
  const int nB   = lane & 15;
  const int kB   = (lane >> 4) * 16;

  int r0 = mr + mrow;       if (r0 >= M) r0 = M - 1;   // clamp loads; stores guarded
  int r1 = mr + 16 + mrow;  if (r1 >= M) r1 = M - 1;
  const __bf16* arow0 = A + (size_t)r0 * DIM + khA;
  const __bf16* arow1 = A + (size_t)r1 * DIM + khA;
  const __bf16* brow0 = BT + (size_t)(nc +  0 + nB) * DIM + kB;
  const __bf16* brow1 = BT + (size_t)(nc + 16 + nB) * DIM + kB;
  const __bf16* brow2 = BT + (size_t)(nc + 32 + nB) * DIM + kB;
  const __bf16* brow3 = BT + (size_t)(nc + 48 + nB) * DIM + kB;

  v8f acc[8] = {};   // acc[t*2 + j], j = M-subtile (0: rows mr.., 1: rows mr+16..)

#pragma unroll
  for (int kb = 0; kb < DIM; kb += 32) {
    v16bf a0 = CAT16(*(const v8bf*)(arow0 + kb), *(const v8bf*)(arow0 + kb + 8));
    v16bf a1 = CAT16(*(const v8bf*)(arow1 + kb), *(const v8bf*)(arow1 + kb + 8));

    v16bf b0 = CAT16(*(const v8bf*)(brow0 + kb), *(const v8bf*)(brow0 + kb + 8));
    acc[0] = __builtin_amdgcn_wmma_f32_16x16x32_bf16(false, a0, false, b0, (short)0, acc[0], false, false);
    acc[1] = __builtin_amdgcn_wmma_f32_16x16x32_bf16(false, a1, false, b0, (short)0, acc[1], false, false);

    v16bf b1 = CAT16(*(const v8bf*)(brow1 + kb), *(const v8bf*)(brow1 + kb + 8));
    acc[2] = __builtin_amdgcn_wmma_f32_16x16x32_bf16(false, a0, false, b1, (short)0, acc[2], false, false);
    acc[3] = __builtin_amdgcn_wmma_f32_16x16x32_bf16(false, a1, false, b1, (short)0, acc[3], false, false);

    v16bf b2 = CAT16(*(const v8bf*)(brow2 + kb), *(const v8bf*)(brow2 + kb + 8));
    acc[4] = __builtin_amdgcn_wmma_f32_16x16x32_bf16(false, a0, false, b2, (short)0, acc[4], false, false);
    acc[5] = __builtin_amdgcn_wmma_f32_16x16x32_bf16(false, a1, false, b2, (short)0, acc[5], false, false);

    v16bf b3 = CAT16(*(const v8bf*)(brow3 + kb), *(const v8bf*)(brow3 + kb + 8));
    acc[6] = __builtin_amdgcn_wmma_f32_16x16x32_bf16(false, a0, false, b3, (short)0, acc[6], false, false);
    acc[7] = __builtin_amdgcn_wmma_f32_16x16x32_bf16(false, a1, false, b3, (short)0, acc[7], false, false);
  }

  // C layout: VGPR r holds (m = (lane>>4)*8 + r, n = lane&15)
#pragma unroll
  for (int t = 0; t < 4; ++t) {
    int colg = nc + t * 16 + (lane & 15);
    float bb = bias[colg];
#pragma unroll
    for (int j = 0; j < 2; ++j) {
#pragma unroll
      for (int r = 0; r < 8; ++r) {
        int m = mr + j * 16 + (lane >> 4) * 8 + r;
        if (m < M) C[(size_t)m * ldc + colg] = acc[t * 2 + j][r] + bb;
      }
    }
  }
}

// ---------------- sparse attention stages ----------------

// rows is sorted: rowptr[i] = lower_bound(rows, i); rowptr[N] = E.
__global__ void build_rowptr(const int* __restrict__ rows, int* __restrict__ rowptr,
                             int Nn, int E) {
  int i = blockIdx.x * blockDim.x + threadIdx.x;
  if (i > Nn) return;
  int lo = 0, hi = E;
  while (lo < hi) { int mid = (lo + hi) >> 1; if (rows[mid] < i) lo = mid + 1; else hi = mid; }
  rowptr[i] = lo;
}

// logits[e*8+h] = (q[rows[e],h,:] . k[cols[e],h,:]) / 8 + att_bias[e,h]
__global__ void edge_logits(const float* __restrict__ QKV, const int* __restrict__ rows,
                            const int* __restrict__ cols, const float* __restrict__ attb,
                            float* __restrict__ logits, int E) {
  int idx = blockIdx.x * blockDim.x + threadIdx.x;
  if (idx >= E * NHEAD) return;
  int e = idx >> 3, h = idx & 7;
  const float4* q = (const float4*)(QKV + (size_t)rows[e] * 1536 + h * DK);
  const float4* k = (const float4*)(QKV + (size_t)cols[e] * 1536 + 512 + h * DK);
  float s = 0.f;
#pragma unroll
  for (int i = 0; i < DK / 4; ++i) {
    float4 a = q[i], b = k[i];
    s += a.x * b.x + a.y * b.y + a.z * b.z + a.w * b.w;
  }
  logits[idx] = s * 0.125f + attb[idx];
}

// per-(node,head) segment softmax, in place
__global__ void seg_softmax(float* __restrict__ logits, const int* __restrict__ rowptr, int Nn) {
  int idx = blockIdx.x * blockDim.x + threadIdx.x;
  if (idx >= Nn * NHEAD) return;
  int i = idx >> 3, h = idx & 7;
  int s = rowptr[i], epos = rowptr[i + 1];
  if (s >= epos) return;
  float m = -3.402823466e38f;
  for (int e = s; e < epos; ++e) m = fmaxf(m, logits[e * 8 + h]);
  float sum = 0.f;
  for (int e = s; e < epos; ++e) sum += expf(logits[e * 8 + h] - m);
  float inv = 1.0f / sum;
  for (int e = s; e < epos; ++e) logits[e * 8 + h] = expf(logits[e * 8 + h] - m) * inv;
}

// y[i,h,:] = sum_e att[e,h] * v[cols[e],h,:]; one 64-thread block per (node,head); emit bf16
__global__ void aggregate(const float* __restrict__ QKV, const float* __restrict__ att,
                          const int* __restrict__ cols, const int* __restrict__ rowptr,
                          __bf16* __restrict__ yb) {
  int b = blockIdx.x;
  int i = b >> 3, h = b & 7;
  int d = threadIdx.x;                 // 0..63
  int s = rowptr[i], epos = rowptr[i + 1];
  float acc = 0.f;
  for (int e = s; e < epos; ++e) {
    float a = att[e * 8 + h];
    acc += a * QKV[(size_t)cols[e] * 1536 + 1024 + h * DK + d];
  }
  yb[(size_t)i * DIM + h * DK + d] = f2bf(acc);
}

// ---------------- launch ----------------

extern "C" void kernel_launch(void* const* d_in, const int* in_sizes, int n_in,
                              void* d_out, int out_size, void* d_ws, size_t ws_size,
                              hipStream_t stream) {
  const float* x    = (const float*)d_in[0];
  const int*   ei   = (const int*)  d_in[1];
  const float* attb = (const float*)d_in[2];
  const float* Wq   = (const float*)d_in[3];
  const float* bq   = (const float*)d_in[4];
  const float* Wk   = (const float*)d_in[5];
  const float* bk   = (const float*)d_in[6];
  const float* Wv   = (const float*)d_in[7];
  const float* bv   = (const float*)d_in[8];
  const float* Wo   = (const float*)d_in[9];
  const float* bo   = (const float*)d_in[10];
  float* out = (float*)d_out;

  const int N = in_sizes[0] / DIM;
  const int E = in_sizes[1] / 2;
  const int* rows = ei;
  const int* cols = ei + E;

  // workspace carving (256B aligned)
  char* ws = (char*)d_ws;
  size_t off = 0;
  auto carve = [&](size_t bytes) -> char* {
    char* p = ws + off;
    off = (off + bytes + 255) & ~(size_t)255;
    return p;
  };
  __bf16* Xb      = (__bf16*)carve((size_t)N * DIM * 2);
  __bf16* WT      = (__bf16*)carve((size_t)2048 * DIM * 2);  // WqT|WkT|WvT|WoT
  float*  biasAll = (float*) carve(2048 * 4);
  float*  QKV     = (float*) carve((size_t)N * 1536 * 4);
  float*  logits  = (float*) carve((size_t)E * NHEAD * 4);   // becomes att in place
  int*    rowptr  = (int*)   carve((size_t)(N + 1) * 4);
  __bf16* yb      = (__bf16*)carve((size_t)N * DIM * 2);
  (void)ws_size; (void)n_in; (void)out_size;

  // 1) convert inputs
  cvt_f32_bf16<<<(N * DIM + 255) / 256, 256, 0, stream>>>(x, Xb, N * DIM);
  prep_weights<<<(2048 * DIM + 255) / 256, 256, 0, stream>>>(
      Wq, Wk, Wv, Wo, bq, bk, bv, bo, WT, biasAll);

  // 2) fused QKV GEMM: [N x 512] x [512 x 1536] -> QKV [N x 1536]
  dim3 g1((N + 127) / 128, 1536 / 64);
  gemm_bf16_wmma<<<g1, 128, 0, stream>>>(Xb, WT, biasAll, QKV, 1536, N);

  // 3) CSR rowptr
  build_rowptr<<<(N + 1 + 255) / 256, 256, 0, stream>>>(rows, rowptr, N, E);

  // 4) edge logits
  edge_logits<<<((size_t)E * NHEAD + 255) / 256, 256, 0, stream>>>(
      QKV, rows, cols, attb, logits, E);

  // 5) segment softmax (in place)
  seg_softmax<<<((size_t)N * NHEAD + 255) / 256, 256, 0, stream>>>(logits, rowptr, N);

  // 6) aggregation -> y (bf16)
  aggregate<<<(size_t)N * NHEAD, DK, 0, stream>>>(QKV, logits, cols, rowptr, yb);

  // 7) output GEMM: [N x 512] x [512 x 512] -> out
  dim3 g2((N + 127) / 128, 512 / 64);
  gemm_bf16_wmma<<<g2, 128, 0, stream>>>(
      yb, WT + (size_t)1536 * DIM, biasAll + 1536, out, DIM, N);
}
